// RAFEE_Encoder_38749194944626
// MI455X (gfx1250) — matrix operand order repaired and checked by reading, not compiled, not run on hardware
//
#include <hip/hip_runtime.h>
#include <math.h>

// ---- problem constants (match reference) ----
#define BB     4
#define LLEN   1024
#define DD     512
#define NLAYER 2
#define NE     8
#define KTOP   4
#define HH     2048
#define BLTOK  (BB * LLEN)   // 4096 tokens

typedef __attribute__((ext_vector_type(16))) __bf16 v16bf;
typedef __attribute__((ext_vector_type(8)))  __bf16 v8bf;
typedef __attribute__((ext_vector_type(8)))  float  v8f;

typedef __attribute__((ext_vector_type(4))) unsigned tdm_u4;
typedef __attribute__((ext_vector_type(8))) int      tdm_i8;
typedef __attribute__((ext_vector_type(4))) int      tdm_i4;

#if defined(__has_builtin)
#if __has_builtin(__builtin_amdgcn_global_load_async_to_lds_b128) && \
    __has_builtin(__builtin_amdgcn_s_wait_asynccnt)
#define HAVE_ASYNC 1
#else
#define HAVE_ASYNC 0
#endif
#if __has_builtin(__builtin_amdgcn_tensor_load_to_lds) && \
    __has_builtin(__builtin_amdgcn_s_wait_tensorcnt)
#define HAVE_TDM 1
#else
#define HAVE_TDM 0
#endif
#else
#define HAVE_ASYNC 0
#define HAVE_TDM 0
#endif

// ============================================================
// generic fp32 -> bf16 conversion (grid-stride)
// ============================================================
__global__ __launch_bounds__(256)
void k_cvt_bf16(const float* __restrict__ src, __bf16* __restrict__ dst, long n) {
  long i = (long)blockIdx.x * blockDim.x + threadIdx.x;
  long stride = (long)gridDim.x * blockDim.x;
  for (; i < n; i += stride) dst[i] = (__bf16)src[i];
}

__global__ __launch_bounds__(256)
void k_init_x(const float* __restrict__ src, float* __restrict__ dstf,
              __bf16* __restrict__ dstb, long n) {
  long i = (long)blockIdx.x * blockDim.x + threadIdx.x;
  long stride = (long)gridDim.x * blockDim.x;
  for (; i < n; i += stride) { float v = src[i]; dstf[i] = v; dstb[i] = (__bf16)v; }
}

// ============================================================
// WMMA bf16 GEMM: C = alpha*A*B (+bias) [relu] [*rowscale] [+=prev]
//   block tile 128x128, 8 waves, each wave 32x64 (2x4 = 8 wmma accums)
//   double-buffered LDS (32 KB), one barrier per K-step
//   A staged via TDM tensor_load_to_lds; B (TRANSB=1) via async->LDS b128
// ============================================================
template <int TRANSB>
__global__ __launch_bounds__(256)
void k_gemm_bf16(const __bf16* __restrict__ A, long strideA, int lda,
                 const __bf16* __restrict__ Bm, long strideB, int ldb,
                 float* outF, long strideC, int ldc,
                 __bf16* outB,
                 const float* __restrict__ bias,
                 const float* __restrict__ rowscale, int rsStride,
                 int K, float alpha, int relu, int accum) {
  __shared__ __bf16 As[2][128][32];
  __shared__ __bf16 Bs[2][128][32];   // [n][k]

  const int bz = blockIdx.z;
  A  += (long)bz * strideA;
  Bm += (long)bz * strideB;
  const long coff = (long)bz * strideC;

  const int m0 = blockIdx.y * 128;
  const int n0 = blockIdx.x * 128;
  const int t    = threadIdx.x;
  const int wave = t >> 5;
  const int lane = t & 31;
  const int wm   = (wave >> 1) * 32;   // 0,32,64,96
  const int wn   = (wave & 1) * 64;    // 0,64
  const int half = lane >> 4;
  const int l16  = lane & 15;

  v8f acc[2][4] = {};

  // ---------- tile staging ----------
  auto stageA = [&](int buf, int k0) {
#if HAVE_TDM
    if (t == 0) {
      // 2-D TDM descriptor: 128x32 bf16 tile, row stride lda (elements)
      unsigned long long ga = (unsigned long long)(uintptr_t)(A + (long)m0 * lda + k0);
      unsigned ldsa = (unsigned)(uintptr_t)(&As[buf][0][0]);
      tdm_u4 g0;
      g0[0] = 1u;                                   // count=1, user mode
      g0[1] = ldsa;                                 // lds_addr
      g0[2] = (unsigned)ga;                         // global_addr[31:0]
      g0[3] = (unsigned)((ga >> 32) & 0x01FFFFFFu) | (2u << 30); // addr[56:32] | type=2
      tdm_i8 g1;
      g1[0] = 1 << 16;                              // data_size = 2 bytes
      g1[1] = (int)0xFFFF0000u;                     // tensor_dim0[15:0] (huge)
      g1[2] = (int)0xFFFF7FFFu;                     // tensor_dim0[31:16]|tensor_dim1[15:0]
      g1[3] = (int)(0x7FFFu | (32u << 16));         // tensor_dim1[31:16] | tile_dim0=32
      g1[4] = 128;                                  // tile_dim1=128, tile_dim2=0
      g1[5] = lda;                                  // tensor_dim0_stride low32 (elements)
      g1[6] = 0;
      g1[7] = 0;
      tdm_i4 gz4 = {0, 0, 0, 0};
      tdm_i8 gz8 = {0, 0, 0, 0, 0, 0, 0, 0};
      __builtin_amdgcn_tensor_load_to_lds(g0, g1, gz4, gz4, gz8, 0);
    }
#else
#pragma unroll
    for (int i = 0; i < 2; ++i) {
      const int c  = t + i * 256;            // 512 chunks of 8 bf16
      const int ml = c >> 2, kl = (c & 3) * 8;
      v8bf av = *reinterpret_cast<const v8bf*>(A + (long)(m0 + ml) * lda + k0 + kl);
      *reinterpret_cast<v8bf*>(&As[buf][ml][kl]) = av;
    }
#endif
  };

  auto stageB = [&](int buf, int k0) {
    if (TRANSB) {
#pragma unroll
      for (int i = 0; i < 2; ++i) {
        const int c  = t + i * 256;
        const int nl = c >> 2, kl = (c & 3) * 8;
#if HAVE_ASYNC
        __builtin_amdgcn_global_load_async_to_lds_b128(
            (tdm_i4*)(&Bs[buf][nl][kl]),
            (tdm_i4*)(Bm + (long)(n0 + nl) * ldb + k0 + kl),
            0, 0);
#else
        v8bf bv = *reinterpret_cast<const v8bf*>(Bm + (long)(n0 + nl) * ldb + k0 + kl);
        *reinterpret_cast<v8bf*>(&Bs[buf][nl][kl]) = bv;
#endif
      }
    } else {
      // B is KxN row-major: transpose-scatter into [n][k]
#pragma unroll
      for (int i = 0; i < 2; ++i) {
        const int c  = t + i * 256;            // 512 chunks
        const int kl = c >> 4, nl = (c & 15) * 8;
        v8bf bv = *reinterpret_cast<const v8bf*>(Bm + (long)(k0 + kl) * ldb + n0 + nl);
#pragma unroll
        for (int u = 0; u < 8; ++u) Bs[buf][nl + u][kl] = bv[u];
      }
    }
  };

  auto waitStages = [&]() {
#if HAVE_ASYNC
    if (TRANSB) __builtin_amdgcn_s_wait_asynccnt(0);
#endif
#if HAVE_TDM
    if (t == 0) __builtin_amdgcn_s_wait_tensorcnt(0);
#endif
  };

  // ---------- software pipeline ----------
  stageA(0, 0);
  stageB(0, 0);

  for (int k0 = 0; k0 < K; k0 += 32) {
    const int buf = (k0 >> 5) & 1;
    waitStages();
    __syncthreads();                 // tile `buf` ready for everyone
    if (k0 + 32 < K) {               // stage next tile into other buffer
      stageA(buf ^ 1, k0 + 32);
      stageB(buf ^ 1, k0 + 32);
    }

    // ---- A fragments (16x32): lanes 0-15 K 0-7/16-23, lanes 16-31 K 8-15/24-31
    v16bf af[2];
#pragma unroll
    for (int i = 0; i < 2; ++i) {
      const __bf16* ar = &As[buf][wm + i * 16 + l16][0];
      v8bf lo = *reinterpret_cast<const v8bf*>(ar + half * 8);
      v8bf hi = *reinterpret_cast<const v8bf*>(ar + 16 + half * 8);
#pragma unroll
      for (int j = 0; j < 8; ++j) { af[i][j] = lo[j]; af[i][8 + j] = hi[j]; }
    }
    // ---- B fragments (32x16): lanes 0-15 K 0-7/16-23, lanes 16-31 K 8-15/24-31
    v16bf bf[4];
#pragma unroll
    for (int j4 = 0; j4 < 4; ++j4) {
      const __bf16* br = &Bs[buf][wn + j4 * 16 + l16][0];
      v8bf p0 = *reinterpret_cast<const v8bf*>(br + half * 8);
      v8bf p1 = *reinterpret_cast<const v8bf*>(br + 16 + half * 8);
#pragma unroll
      for (int j = 0; j < 8; ++j) { bf[j4][j] = p0[j]; bf[j4][8 + j] = p1[j]; }
    }
#pragma unroll
    for (int i = 0; i < 2; ++i)
#pragma unroll
      for (int j4 = 0; j4 < 4; ++j4)
        acc[i][j4] = __builtin_amdgcn_wmma_f32_16x16x32_bf16(
            false, af[i], false, bf[j4], (short)0, acc[i][j4], false, false);
  }

  // ---- epilogue: C/D layout -> row = r + 8*half, col = lane&15 ----
#pragma unroll
  for (int i = 0; i < 2; ++i) {
    const int mb = m0 + wm + i * 16 + half * 8;
#pragma unroll
    for (int r = 0; r < 8; ++r) {
      const int m = mb + r;
      const float rs = rowscale ? rowscale[(long)m * rsStride] : 1.f;
#pragma unroll
      for (int j4 = 0; j4 < 4; ++j4) {
        const int ncj = n0 + wn + j4 * 16 + l16;
        float v = alpha * acc[i][j4][r];
        if (bias) v += bias[ncj];
        if (relu) v = fmaxf(v, 0.f);
        v *= rs;
        const long idx = coff + (long)m * ldc + ncj;
        float vv = v;
        if (accum && outF) vv += outF[idx];
        if (outF) outF[idx] = vv;
        if (outB) outB[idx] = (__bf16)vv;
      }
    }
  }
}

// ============================================================
// RoPE: f = norm_coord @ wr[l]; rotate q,k pairs; emit bf16
// ============================================================
__global__ __launch_bounds__(256)
void k_rope(const float* __restrict__ qf, const float* __restrict__ kf,
            const float* __restrict__ nc, const float* __restrict__ wr,
            __bf16* __restrict__ qb, __bf16* __restrict__ kb) {
  const int idx = blockIdx.x * 256 + threadIdx.x;
  const int j = idx & 255;
  const int tkn = idx >> 8;
  if (tkn >= BLTOK) return;
  const float c0 = nc[2 * tkn], c1 = nc[2 * tkn + 1];
  const float f = c0 * wr[j] + c1 * wr[256 + j];
  const float cf = cosf(f), sf = sinf(f);
  const long base = (long)tkn * DD + 2 * j;
  { float re = qf[base], im = qf[base + 1];
    qb[base]     = (__bf16)(re * cf - im * sf);
    qb[base + 1] = (__bf16)(re * sf + im * cf); }
  { float re = kf[base], im = kf[base + 1];
    kb[base]     = (__bf16)(re * cf - im * sf);
    kb[base + 1] = (__bf16)(re * sf + im * cf); }
}

// ============================================================
// causal softmax over a score row; writes normalized P in bf16
// ============================================================
__global__ __launch_bounds__(256)
void k_softmax(const float* __restrict__ S, __bf16* __restrict__ P) {
  const int q = blockIdx.x, b = blockIdx.y, t = threadIdx.x;
  const float* row = S + ((long)b * LLEN + q) * LLEN;
  __bf16* prow = P + ((long)b * LLEN + q) * LLEN;
  const int valid = q + 1;
  __shared__ float sm[256];

  float mx = -INFINITY;
  for (int k = t; k < valid; k += 256) mx = fmaxf(mx, row[k]);
  sm[t] = mx; __syncthreads();
  for (int s = 128; s > 0; s >>= 1) { if (t < s) sm[t] = fmaxf(sm[t], sm[t + s]); __syncthreads(); }
  mx = sm[0]; __syncthreads();

  float sum = 0.f;
  for (int k = t; k < valid; k += 256) sum += expf(row[k] - mx);
  sm[t] = sum; __syncthreads();
  for (int s = 128; s > 0; s >>= 1) { if (t < s) sm[t] += sm[t + s]; __syncthreads(); }
  const float inv = 1.f / sm[0];

  for (int k = t; k < LLEN; k += 256)
    prow[k] = (__bf16)((k < valid) ? expf(row[k] - mx) * inv : 0.f);
}

// ============================================================
// residual add + LayerNorm (D=512, block 256, 2 elems/thread)
// ============================================================
__global__ __launch_bounds__(256)
void k_resid_ln(const float* __restrict__ a, const float* __restrict__ r,
                const float* __restrict__ g, const float* __restrict__ bta,
                float* __restrict__ outF, __bf16* __restrict__ outB) {
  const int tkn = blockIdx.x, t = threadIdx.x;
  const long base = (long)tkn * DD;
  float v0 = a[base + t]       + r[base + t];
  float v1 = a[base + t + 256] + r[base + t + 256];
  __shared__ float s1[256], s2[256];
  s1[t] = v0 + v1; s2[t] = v0 * v0 + v1 * v1;
  __syncthreads();
  for (int s = 128; s > 0; s >>= 1) {
    if (t < s) { s1[t] += s1[t + s]; s2[t] += s2[t + s]; }
    __syncthreads();
  }
  const float mean = s1[0] * (1.f / DD);
  const float var  = s2[0] * (1.f / DD) - mean * mean;
  const float inv  = rsqrtf(var + 1e-5f);
  const float y0 = (v0 - mean) * inv * g[t]       + bta[t];
  const float y1 = (v1 - mean) * inv * g[t + 256] + bta[t + 256];
  outF[base + t] = y0;         outF[base + t + 256] = y1;
  outB[base + t] = (__bf16)y0; outB[base + t + 256] = (__bf16)y1;
}

// ============================================================
// router: logits/noise matvecs (D x 8), noisy top-4, gate softmax
// ============================================================
__device__ inline unsigned hash_u32(unsigned x) {
  x ^= x >> 16; x *= 0x7feb352dU; x ^= x >> 15; x *= 0x846ca68bU; x ^= x >> 16;
  return x;
}
__device__ inline float gauss_hash(unsigned s) {
  unsigned a = hash_u32(s * 2654435761u + 1u);
  unsigned b = hash_u32(a ^ 0x9e3779b9u);
  float u1 = ((float)(a >> 8) + 0.5f) * (1.f / 16777216.f);
  float u2 = ((float)(b >> 8)) * (1.f / 16777216.f);
  return sqrtf(-2.f * logf(u1)) * cosf(6.28318530718f * u2);
}

__global__ __launch_bounds__(256)
void k_router(const float* __restrict__ x1, const float* __restrict__ rtw,
              const float* __restrict__ rtb, const float* __restrict__ nzw,
              const float* __restrict__ nzb, float* __restrict__ gate, int layer) {
  __shared__ float swr[DD * NE];
  __shared__ float swn[DD * NE];
  for (int i = threadIdx.x; i < DD * NE; i += 256) { swr[i] = rtw[i]; swn[i] = nzw[i]; }
  __syncthreads();

  const int tkn = blockIdx.x * 256 + threadIdx.x;
  const float* xr = x1 + (long)tkn * DD;
  float lg[NE], nz[NE];
#pragma unroll
  for (int e = 0; e < NE; ++e) { lg[e] = rtb[e]; nz[e] = nzb[e]; }
  for (int k = 0; k < DD; ++k) {
    const float xv = xr[k];
#pragma unroll
    for (int e = 0; e < NE; ++e) {
      lg[e] += xv * swr[k * NE + e];
      nz[e] += xv * swn[k * NE + e];
    }
  }
  float noisy[NE]; bool sel[NE];
#pragma unroll
  for (int e = 0; e < NE; ++e) {
    const float sp = (nz[e] > 20.f) ? nz[e] : log1pf(expf(nz[e]));
    const unsigned seed = ((unsigned)layer * BLTOK + (unsigned)tkn) * NE + e;
    noisy[e] = lg[e] + gauss_hash(seed) * sp;
    sel[e] = false;
  }
  for (int it = 0; it < KTOP; ++it) {
    int best = 0; float bv = -INFINITY;
#pragma unroll
    for (int e = 0; e < NE; ++e)
      if (!sel[e] && noisy[e] > bv) { bv = noisy[e]; best = e; }
    sel[best] = true;
  }
  float mx = -INFINITY;
#pragma unroll
  for (int e = 0; e < NE; ++e) if (sel[e]) mx = fmaxf(mx, noisy[e]);
  float s = 0.f, ge[NE];
#pragma unroll
  for (int e = 0; e < NE; ++e) { ge[e] = sel[e] ? expf(noisy[e] - mx) : 0.f; s += ge[e]; }
  const float inv = 1.f / s;
#pragma unroll
  for (int e = 0; e < NE; ++e) gate[(long)tkn * NE + e] = ge[e] * inv;
}

// ============================================================
// host-side orchestration
// ============================================================
extern "C" void kernel_launch(void* const* d_in, const int* in_sizes, int n_in,
                              void* d_out, int out_size, void* d_ws, size_t ws_size,
                              hipStream_t stream) {
  const float* x    = (const float*)d_in[0];
  const float* nc   = (const float*)d_in[1];
  const float* wq   = (const float*)d_in[4];
  const float* bq   = (const float*)d_in[5];
  const float* wk   = (const float*)d_in[6];
  const float* bk   = (const float*)d_in[7];
  const float* wv   = (const float*)d_in[8];
  const float* bv   = (const float*)d_in[9];
  const float* wr   = (const float*)d_in[10];
  const float* ln1g = (const float*)d_in[11];
  const float* ln1b = (const float*)d_in[12];
  const float* rtw  = (const float*)d_in[13];
  const float* rtb  = (const float*)d_in[14];
  const float* nzw  = (const float*)d_in[15];
  const float* nzb  = (const float*)d_in[16];
  const float* ew1  = (const float*)d_in[17];
  const float* eb1  = (const float*)d_in[18];
  const float* ew2  = (const float*)d_in[19];
  const float* eb2  = (const float*)d_in[20];
  const float* ln2g = (const float*)d_in[21];
  const float* ln2b = (const float*)d_in[22];

  char* base = (char*)d_ws;
  size_t off = 0;
  auto alloc = [&](size_t bytes) -> void* {
    void* p = base + off;
    off = (off + bytes + 255) & ~(size_t)255;
    return p;
  };

  __bf16* wq_b  = (__bf16*)alloc((size_t)NLAYER * DD * DD * 2);
  __bf16* wk_b  = (__bf16*)alloc((size_t)NLAYER * DD * DD * 2);
  __bf16* wv_b  = (__bf16*)alloc((size_t)NLAYER * DD * DD * 2);
  __bf16* ew1_b = (__bf16*)alloc((size_t)NLAYER * NE * DD * HH * 2);
  __bf16* ew2_b = (__bf16*)alloc((size_t)NLAYER * NE * HH * DD * 2);
  float*  xcur  = (float*) alloc((size_t)BLTOK * DD * 4);
  float*  x1    = (float*) alloc((size_t)BLTOK * DD * 4);
  __bf16* xb    = (__bf16*)alloc((size_t)BLTOK * DD * 2);
  __bf16* x1b   = (__bf16*)alloc((size_t)BLTOK * DD * 2);
  float*  qf    = (float*) alloc((size_t)BLTOK * DD * 4);   // reused as attn out
  float*  kf    = (float*) alloc((size_t)BLTOK * DD * 4);   // reused as moe acc
  __bf16* qb    = (__bf16*)alloc((size_t)BLTOK * DD * 2);
  __bf16* kb    = (__bf16*)alloc((size_t)BLTOK * DD * 2);
  __bf16* vb    = (__bf16*)alloc((size_t)BLTOK * DD * 2);
  float*  scores= (float*) alloc((size_t)BB * LLEN * LLEN * 4); // reused as hb
  __bf16* P     = (__bf16*)alloc((size_t)BB * LLEN * LLEN * 2);
  float*  gate  = (float*) alloc((size_t)BLTOK * NE * 4);
  float*  attn  = qf;
  float*  moe   = kf;
  __bf16* hb    = (__bf16*)scores;

  const float inv_sqrt_d = 1.0f / sqrtf((float)DD);

  // ---- weight conversion (once per call) ----
  k_cvt_bf16<<<2048, 256, 0, stream>>>(wq,  wq_b,  (long)NLAYER * DD * DD);
  k_cvt_bf16<<<2048, 256, 0, stream>>>(wk,  wk_b,  (long)NLAYER * DD * DD);
  k_cvt_bf16<<<2048, 256, 0, stream>>>(wv,  wv_b,  (long)NLAYER * DD * DD);
  k_cvt_bf16<<<4096, 256, 0, stream>>>(ew1, ew1_b, (long)NLAYER * NE * DD * HH);
  k_cvt_bf16<<<4096, 256, 0, stream>>>(ew2, ew2_b, (long)NLAYER * NE * HH * DD);
  k_init_x <<<2048, 256, 0, stream>>>(x, xcur, xb, (long)BLTOK * DD);

  for (int l = 0; l < NLAYER; ++l) {
    const long wOff = (long)l * DD * DD;
    // ---- QKV projections: [4096x512] x [512x512] ----
    dim3 gq(DD / 128, BLTOK / 128, 1);
    k_gemm_bf16<0><<<gq, 256, 0, stream>>>(xb, 0, DD, wq_b + wOff, 0, DD,
                                           qf, 0, DD, nullptr, bq + l * DD,
                                           nullptr, 0, DD, 1.f, 0, 0);
    k_gemm_bf16<0><<<gq, 256, 0, stream>>>(xb, 0, DD, wk_b + wOff, 0, DD,
                                           kf, 0, DD, nullptr, bk + l * DD,
                                           nullptr, 0, DD, 1.f, 0, 0);
    k_gemm_bf16<0><<<gq, 256, 0, stream>>>(xb, 0, DD, wv_b + wOff, 0, DD,
                                           nullptr, 0, DD, vb, bv + l * DD,
                                           nullptr, 0, DD, 1.f, 0, 0);
    // ---- RoPE ----
    k_rope<<<(BLTOK * 256) / 256, 256, 0, stream>>>(qf, kf, nc, wr + (long)l * DD, qb, kb);
    // ---- scores = q k^T / sqrt(D), batched, transB ----
    dim3 gs(LLEN / 128, LLEN / 128, BB);
    k_gemm_bf16<1><<<gs, 256, 0, stream>>>(qb, (long)LLEN * DD, DD,
                                           kb, (long)LLEN * DD, DD,
                                           scores, (long)LLEN * LLEN, LLEN, nullptr,
                                           nullptr, nullptr, 0, DD, inv_sqrt_d, 0, 0);
    // ---- causal softmax -> P (bf16) ----
    k_softmax<<<dim3(LLEN, BB), 256, 0, stream>>>(scores, P);
    // ---- attn = P @ V, batched ----
    dim3 ga(DD / 128, LLEN / 128, BB);
    k_gemm_bf16<0><<<ga, 256, 0, stream>>>(P, (long)LLEN * LLEN, LLEN,
                                           vb, (long)LLEN * DD, DD,
                                           attn, (long)LLEN * DD, DD, nullptr,
                                           nullptr, nullptr, 0, LLEN, 1.f, 0, 0);
    // ---- x1 = LN(xcur + attn) ----
    k_resid_ln<<<BLTOK, 256, 0, stream>>>(xcur, attn, ln1g + l * DD, ln1b + l * DD, x1, x1b);
    // ---- router -> gate ----
    k_router<<<BLTOK / 256, 256, 0, stream>>>(x1, rtw + (long)l * DD * NE, rtb + l * NE,
                                              nzw + (long)l * DD * NE, nzb + l * NE, gate, l);
    // ---- MoE: dense dispatch, gate==0 for non-top-k ----
    for (int e = 0; e < NE; ++e) {
      const long w1Off = ((long)l * NE + e) * DD * HH;
      const long w2Off = ((long)l * NE + e) * HH * DD;
      dim3 g1(HH / 128, BLTOK / 128, 1);
      k_gemm_bf16<0><<<g1, 256, 0, stream>>>(x1b, 0, DD, ew1_b + w1Off, 0, HH,
                                             nullptr, 0, HH, hb,
                                             eb1 + ((long)l * NE + e) * HH,
                                             nullptr, 0, DD, 1.f, /*relu*/1, 0);
      dim3 g2(DD / 128, BLTOK / 128, 1);
      k_gemm_bf16<0><<<g2, 256, 0, stream>>>(hb, 0, HH, ew2_b + w2Off, 0, DD,
                                             moe, 0, DD, nullptr,
                                             eb2 + ((long)l * NE + e) * DD,
                                             gate + e, NE, HH, 1.f, 0, /*accum*/(e > 0));
    }
    // ---- x = LN(x1 + moe); last layer writes d_out ----
    float* outPtr = (l == NLAYER - 1) ? (float*)d_out : xcur;
    k_resid_ln<<<BLTOK, 256, 0, stream>>>(x1, moe, ln2g + l * DD, ln2b + l * DD, outPtr, xb);
  }
  (void)in_sizes; (void)n_in; (void)out_size; (void)ws_size;
}
